// LRIUpBlock_5884105195891
// MI455X (gfx1250) — matrix operand order repaired and checked by reading, not compiled, not run on hardware
//
#include <hip/hip_runtime.h>

// ---------------------------------------------------------------------------
// LRI bispectrum up-block for MI455X (gfx1250, wave32, WMMA f16->f32).
//
// Pipeline (all on `stream`):
//   build_wk<128>  : w_t -> swizzled f16 B-matrix for the transposed conv
//   build_wk<128>  : w1  -> swizzled f16 B-matrix for conv1
//   build_wk<64>   : w2  -> swizzled f16 B-matrix for conv2
//   concat_skip    : skip(f32) -> hcat[..., 64:128] (f16)
//   conv_bispec<T> : x -> hcat[..., 0:64]     (implicit-GEMM transposed conv)
//   conv_bispec    : hcat -> h2               (stride-1 conv, Cin=128)
//   conv_bispec    : h2 -> d_out (f32)        (stride-1 conv, Cin=64)
// ---------------------------------------------------------------------------

typedef __attribute__((ext_vector_type(16))) _Float16 v16h;
typedef __attribute__((ext_vector_type(8)))  _Float16 v8h;
typedef __attribute__((ext_vector_type(8)))  float    v8f;

#define HOUT 128
#define WOUT 128

// -------------------- circular-harmonic atoms (ksize=3, NH=4) --------------
// Positions (row-major 3x3): ring id (0=center,1=edge,2=corner) and angle in
// units of 45deg.  Norm: ring0 -> 1 element, rings 1/2 -> 4 elements (nrm=2).
__device__ __forceinline__ float atom_val(int n, int r, int tap, int imag) {
  const int ring[9] = {2, 1, 2, 1, 0, 1, 2, 1, 2};
  const int koct[9] = {5, 6, 7, 4, 0, 0, 3, 2, 1};   // atan2 angle / 45deg mod 8
  if (ring[tap] != r) return 0.f;
  if (n > 0 && r == 0) return 0.f;                   // zero atom (rv < eps)
  const float R2 = 0.70710678118654752f;
  const float c8[8] = {1.f, R2, 0.f, -R2, -1.f, -R2, 0.f, R2};
  const float s8[8] = {0.f, R2, 1.f, R2, 0.f, -R2, -1.f, -R2};
  int a = (n * koct[tap]) & 7;
  float v = imag ? s8[a] : c8[a];
  return (r == 0) ? v : v * 0.5f;
}

// -------------------- weight build: w (NH,3,CI,64) -> WMMA B layout --------
// Conv kernel channel layout (matches reference concat): nout in [0,512):
//   nout < 256 : real part, nout = n*64 + co ;  nout >= 256 : imag part.
// B layout this code uses for v_wmma_f32_16x16x32_f16 (32x16 K x N, wave32, k order as A):
//   lane = (krem & 16) + (nout & 15), element e = krem & 15, krem = k % 32.
// Stored as Bbuf[ntile][kstep][lane][16] f16 -> each lane reads 2x b128.
template<int CI>
__global__ __launch_bounds__(256) void build_wk(const float* __restrict__ w,
                                                _Float16* __restrict__ B) {
  const int total = 9 * CI * 512;
  int idx = blockIdx.x * 256 + threadIdx.x;
  if (idx >= total) return;
  int nout = idx & 511;
  int k    = idx >> 9;                // k = (ky*3+kx)*CI + ci
  int ci   = k % CI;
  int tap  = k / CI;
  int part = nout >> 8;
  int n    = (nout >> 6) & 3;
  int co   = nout & 63;
  float acc = 0.f;
#pragma unroll
  for (int r = 0; r < 3; ++r)
    acc += atom_val(n, r, tap, part) * w[((n * 3 + r) * CI + ci) * 64 + co];
  const int KST = (9 * CI) / 32;
  int ks = k >> 5, krem = k & 31;
  int lane  = (krem & 16) + (nout & 15);
  int e     = krem & 15;
  int ntile = nout >> 4;
  B[(((size_t)ntile * KST + ks) * 32 + lane) * 16 + e] = (_Float16)acc;
}

// -------------------- skip concat: f32 -> f16 into hcat[...,64:128] --------
__global__ __launch_bounds__(256) void concat_skip(const float* __restrict__ skip,
                                                   _Float16* __restrict__ hcat) {
  size_t idx = (size_t)blockIdx.x * 256 + threadIdx.x;  // 8*128*128*64 total
  size_t px = idx >> 6;
  int c = (int)(idx & 63);
  hcat[px * 128 + 64 + c] = (_Float16)skip[idx];
}

// -------------------- fused implicit-GEMM conv + bispectrum ---------------
// One workgroup: 32 output pixels (one row segment) x 512 channels.
// 8 waves, wave w owns N columns [w*64, w*64+64): 2 M-tiles x 4 N-tiles.
template<int CI, int HIN, bool TRANSPOSE, typename TIN, bool OUTF32, int OCSTR>
__global__ __launch_bounds__(256)
void conv_bispec(const TIN* __restrict__ in, const _Float16* __restrict__ Bw,
                 const float* __restrict__ bias, _Float16* __restrict__ outH,
                 float* __restrict__ outF) {
  constexpr int CIP = CI + 8;                 // LDS pad: kills bank conflicts
  constexpr int KST = (9 * CI) / 32;
  constexpr int FSTRIDE = 516;                // 512 + 4 f32 pad
  constexpr int PATCH_B = 3 * 34 * CIP * 2;
  constexpr int FBUF_B  = 16 * FSTRIDE * 4;
  constexpr int SMEM_B  = (PATCH_B > FBUF_B) ? PATCH_B : FBUF_B;
  __shared__ __align__(16) char smem[SMEM_B];
  _Float16* patch = (_Float16*)smem;          // [3][34][CIP], dead after GEMM
  float*    fbuf  = (float*)smem;             // [16][FSTRIDE], after GEMM

  const int tid = threadIdx.x;
  const int x0  = blockIdx.x * 32;
  const int oy  = blockIdx.y;
  const int b   = blockIdx.z;

  // ---- stage im2col patch (zero-dilated for the transposed conv) ----
  for (int idx = tid; idx < 3 * 34 * CI; idx += 256) {
    int ci = idx % CI;
    int t = idx / CI;
    int col = t % 34;
    int row = t / 34;
    float v = 0.f;
    if (TRANSPOSE) {
      // conv_transpose k=3 s=2 SAME: pad (2,1), lhs dilation 2, no flip.
      int dy = oy + row - 2, dx = x0 + col - 2;
      if (dy >= 0 && dx >= 0 && !(dy & 1) && !(dx & 1)) {
        int iy = dy >> 1, ix = dx >> 1;
        if (iy < HIN && ix < HIN)
          v = (float)in[(((size_t)b * HIN + iy) * HIN + ix) * CI + ci];
      }
    } else {
      int iy = oy - 1 + row, ix = x0 - 1 + col;
      if (iy >= 0 && iy < HIN && ix >= 0 && ix < HIN)
        v = (float)in[(((size_t)b * HIN + iy) * HIN + ix) * CI + ci];
    }
    patch[(row * 34 + col) * CIP + ci] = (_Float16)v;
  }
  __syncthreads();

  const int lane = tid & 31;
  const int wave = tid >> 5;
  const int mrow = lane & 15;
  const int hsel = (lane & 16) >> 1;          // 0 | 8 : C row half
  v8f acc[2][4] = {};

#pragma unroll 2
  for (int ks = 0; ks < KST; ++ks) {
    const int kb  = ks * 32;
    const int tap = kb / CI;                  // 32 | CI -> one tap per k-step
    const int kx = tap % 3, ky = tap / 3;
    const int ci = (kb % CI) + (lane & 16);
    // A fragments: lane holds K {base..base+15} -> 2x b128
    v16h a[2];
#pragma unroll
    for (int mt = 0; mt < 2; ++mt) {
      const _Float16* p = &patch[(ky * 34 + (mt * 16 + mrow + kx)) * CIP + ci];
      union { v16h v; v8h h[2]; } ua;
      ua.h[0] = *(const v8h*)p;
      ua.h[1] = *(const v8h*)(p + 8);
      a[mt] = ua.v;
    }
    const v8h* bp = (const v8h*)Bw + ((size_t)(wave * 4) * KST + ks) * 64 + lane * 2;
#pragma unroll
    for (int nt = 0; nt < 4; ++nt) {
      union { v16h v; v8h h[2]; } ub;
      const v8h* q = bp + (size_t)nt * (KST * 64);
      ub.h[0] = q[0];
      ub.h[1] = q[1];
      acc[0][nt] = __builtin_amdgcn_wmma_f32_16x16x32_f16(
          false, a[0], false, ub.v, (short)0, acc[0][nt], false, false);
      acc[1][nt] = __builtin_amdgcn_wmma_f32_16x16x32_f16(
          false, a[1], false, ub.v, (short)0, acc[1][nt], false, false);
    }
  }
  __syncthreads();   // patch dead; fbuf takes over the LDS region

  // ---- bispectrum + bias + relu, 16 pixels at a time ----
  for (int mt = 0; mt < 2; ++mt) {
#pragma unroll
    for (int nt = 0; nt < 4; ++nt) {
      int chan = wave * 64 + nt * 16 + mrow;
#pragma unroll
      for (int r = 0; r < 8; ++r)            // C/D layout: M = r + (lane>=16)*8
        fbuf[(r + hsel) * FSTRIDE + chan] = acc[mt][nt][r];
    }
    __syncthreads();
    for (int i = tid; i < 1024; i += 256) {  // 16 px x 64 co
      int p = i >> 6, co = i & 63;
      float fr[4], fi[4];
#pragma unroll
      for (int n = 0; n < 4; ++n) {
        fr[n] = fbuf[p * FSTRIDE + n * 64 + co];
        fi[n] = fbuf[p * FSTRIDE + 256 + n * 64 + co];
      }
      float s = bias[co];
      // pairs (n1,n2): (0,0)(0,1)(0,2)(0,3)(1,1)(1,2); Re(f1*f2*conj(f12))
#define BIS(A, B2, C)                                             \
      { float pr = fr[A]*fr[B2] - fi[A]*fi[B2];                   \
        float pi = fr[A]*fi[B2] + fi[A]*fr[B2];                   \
        s += pr * fr[C] + pi * fi[C]; }
      BIS(0, 0, 0) BIS(0, 1, 1) BIS(0, 2, 2) BIS(0, 3, 3) BIS(1, 1, 2) BIS(1, 2, 3)
#undef BIS
      s = fmaxf(s, 0.f);
      int ox = x0 + mt * 16 + p;
      size_t o = ((size_t)b * HOUT + oy) * WOUT + ox;
      if (OUTF32) outF[o * 64 + co] = s;
      else        outH[o * OCSTR + co] = (_Float16)s;
    }
    __syncthreads();
  }
}

// -------------------- workspace layout (bytes) -----------------------------
//   0         : Kt  (3*3*128*512 f16) = 1,179,648
//   1,179,648 : K1                      1,179,648
//   2,359,296 : K2  (3*3*64*512  f16) =   589,824
//   2,949,120 : hcat (8*128*128*128 f16) = 33,554,432
//  36,503,552 : h2   (8*128*128*64  f16) = 16,777,216
//  total ~ 50.8 MB

extern "C" void kernel_launch(void* const* d_in, const int* in_sizes, int n_in,
                              void* d_out, int out_size, void* d_ws, size_t ws_size,
                              hipStream_t stream) {
  const float* x    = (const float*)d_in[0];
  const float* skip = (const float*)d_in[1];
  const float* w_t  = (const float*)d_in[2];
  const float* b_t  = (const float*)d_in[3];
  const float* w1   = (const float*)d_in[4];
  const float* b1   = (const float*)d_in[5];
  const float* w2   = (const float*)d_in[6];
  const float* b2   = (const float*)d_in[7];
  float* out = (float*)d_out;

  char* ws = (char*)d_ws;
  _Float16* Kt   = (_Float16*)(ws);
  _Float16* K1   = (_Float16*)(ws + 1179648);
  _Float16* K2   = (_Float16*)(ws + 2359296);
  _Float16* hcat = (_Float16*)(ws + 2949120);
  _Float16* h2   = (_Float16*)(ws + 36503552);

  build_wk<128><<<2304, 256, 0, stream>>>(w_t, Kt);
  build_wk<128><<<2304, 256, 0, stream>>>(w1, K1);
  build_wk<64><<<1152, 256, 0, stream>>>(w2, K2);
  concat_skip<<<32768, 256, 0, stream>>>(skip, hcat);

  dim3 grid(WOUT / 32, HOUT, 8);
  // stage T: transposed conv from x (64x64, Cin=128) into hcat[...,0:64]
  conv_bispec<128, 64, true, float, false, 128>
      <<<grid, 256, 0, stream>>>(x, Kt, b_t, hcat, nullptr);
  // stage 1: conv on hcat (Cin=128) -> h2 (f16)
  conv_bispec<128, 128, false, _Float16, false, 64>
      <<<grid, 256, 0, stream>>>(hcat, K1, b1, h2, nullptr);
  // stage 2: conv on h2 (Cin=64) -> d_out (f32)
  conv_bispec<64, 128, false, _Float16, true, 64>
      <<<grid, 256, 0, stream>>>(h2, K2, b2, nullptr, out);
}
